// MultiHeadedAttention_42107859370957
// MI455X (gfx1250) — matrix operand order repaired and checked by reading, not compiled, not run on hardware
//
#include <hip/hip_runtime.h>
#include <hip/hip_bf16.h>
#include <math.h>

// ---------------------------------------------------------------------------
// MultiHeadedAttention (2-stage image attention) for MI455X / gfx1250.
// All matmuls via v_wmma_f32_16x16x32_bf16 (f32 accumulate), softmax in f32.
// GEMM: 128x128 block tile, BK=64, double-buffered LDS staged with
// global_load_async_to_lds_b128 (ASYNCcnt) when available.
// ---------------------------------------------------------------------------

typedef __bf16 bf16;
typedef __attribute__((ext_vector_type(16))) __bf16 bf16x16;
typedef __attribute__((ext_vector_type(8)))  float   v8f;

union FragAB { bf16x16 v; uint4 q[2]; };

#define HEADS  8
#define DMODEL 4096
#define DK     512
#define BSZ    4
#define SQL    2048
#define SKL    512

// ---------------- gfx1250 async global->LDS staging helpers ----------------
#if __has_builtin(__builtin_amdgcn_global_load_async_to_lds_b128)
#define HAVE_ASYNC_LDS 1
typedef int v4i __attribute__((vector_size(16)));
typedef __attribute__((address_space(1))) v4i as1_v4i;   // "__device__" in diags
typedef __attribute__((address_space(3))) v4i as3_v4i;   // "__shared__" in diags
#else
#define HAVE_ASYNC_LDS 0
#endif

__device__ __forceinline__ void cp16_g2l(const bf16* __restrict__ g, bf16* l) {
#if HAVE_ASYNC_LDS
  __builtin_amdgcn_global_load_async_to_lds_b128((as1_v4i*)g, (as3_v4i*)l, 0, 0);
#else
  *(uint4*)l = *(const uint4*)g;   // sync fallback: global->VGPR->ds_store
#endif
}

template <int N> __device__ __forceinline__ void wait_async_le() {
#if HAVE_ASYNC_LDS
#if __has_builtin(__builtin_amdgcn_s_wait_asynccnt)
  __builtin_amdgcn_s_wait_asynccnt(N);
#else
  asm volatile("s_wait_asynccnt %0" ::"i"(N) : "memory");
#endif
#endif
}

// -------------------- f32 -> bf16 elementwise (x4 vectorized) --------------
__global__ void __launch_bounds__(256) k_cvt_bf16(const float* __restrict__ in,
                                                  bf16* __restrict__ out, long n4) {
  long i = (long)blockIdx.x * 256 + threadIdx.x;
  if (i >= n4) return;
  const float4 f = ((const float4*)in)[i];
  union { bf16 h[4]; uint2 u; } p;
  p.h[0] = (bf16)f.x; p.h[1] = (bf16)f.y; p.h[2] = (bf16)f.z; p.h[3] = (bf16)f.w;
  ((uint2*)out)[i] = p.u;
}

// ---------- f32 -> bf16 transposing convert: Wt[k][n] = W[n][k] ------------
__global__ void __launch_bounds__(256) k_cvtT_bf16(const float* __restrict__ W,
                                                   bf16* __restrict__ Wt, int dim) {
  __shared__ bf16 tile[64][65];
  const int tx = threadIdx.x & 63;
  const int ty = threadIdx.x >> 6;            // 0..3
  const long cb = (long)blockIdx.x * 64;      // k block
  const long rb = (long)blockIdx.y * 64;      // n block
#pragma unroll
  for (int i = 0; i < 16; ++i) {
    const int rr = i * 4 + ty;
    tile[rr][tx] = (bf16)W[(rb + rr) * dim + cb + tx];
  }
  __syncthreads();
#pragma unroll
  for (int i = 0; i < 16; ++i) {
    const int rr = i * 4 + ty;
    Wt[(cb + rr) * dim + rb + tx] = tile[tx][rr];
  }
}

// -------------------- row softmax over 512, f32 -> bf16 --------------------
__global__ void __launch_bounds__(256) k_softmax512(const float* __restrict__ in,
                                                    bf16* __restrict__ out) {
  __shared__ float red[256];
  const long row = blockIdx.x;
  const float* r = in + row * 512;
  bf16* o = out + row * 512;
  const int t = threadIdx.x;
  const float a = r[t], b = r[t + 256];
  red[t] = fmaxf(a, b);
  __syncthreads();
  for (int s = 128; s > 0; s >>= 1) {
    if (t < s) red[t] = fmaxf(red[t], red[t + s]);
    __syncthreads();
  }
  const float mx = red[0];
  __syncthreads();
  const float ea = __expf(a - mx), eb = __expf(b - mx);
  red[t] = ea + eb;
  __syncthreads();
  for (int s = 128; s > 0; s >>= 1) {
    if (t < s) red[t] += red[t + s];
    __syncthreads();
  }
  const float inv = 1.f / red[0];
  o[t]       = (bf16)(ea * inv);
  o[t + 256] = (bf16)(eb * inv);
}

// ---------------------------------------------------------------------------
// Batched bf16 WMMA GEMM:  C[z] = alpha * A[z] (MxK) @ B[z] (KxN) + bias[n]
// 128x128 block, 8 waves, wave tile 64x32 (4x2 WMMA 16x16x32 accumulators),
// BK=64 double-buffered LDS staged via async-to-LDS loads.
// Output scatter: oi = (z/zdiv)*osz1 + (z%zdiv)*osz2 + m*osm
//                    + (n/nbDiv)*osnb + (n%nbDiv)*osn
// All dims assumed multiples of tile sizes (true for this problem).
// ---------------------------------------------------------------------------
#define LDA_T 72    // 64 K-halfs + 8 pad  (144B row stride, 16B aligned)
#define LDB_T 136   // 128 N-halfs + 8 pad (272B row stride, 16B aligned)

__global__ void __launch_bounds__(256) k_wmma_gemm(
    const bf16* __restrict__ A, long lda, long aBatch,
    const bf16* __restrict__ B, long ldb, long bBatch,
    const float* __restrict__ bias, float alpha,
    float* __restrict__ Cf, bf16* __restrict__ Cb,
    int K,
    int zdiv, long osz1, long osz2, long osm, int nbDiv, long osnb, long osn) {
  __shared__ __align__(16) bf16 ldsA[2 * 128 * LDA_T];  // 36.9 KB
  __shared__ __align__(16) bf16 ldsB[2 * 64 * LDB_T];   // 34.8 KB

  const int  tid = threadIdx.x;
  const int  z   = blockIdx.z;
  const long m0  = (long)blockIdx.y * 128;
  const long n0  = (long)blockIdx.x * 128;

  const bf16* Ab = A + (long)z * aBatch + m0 * lda;
  const bf16* Bb = B + (long)z * bBatch + n0;

  const int wave = tid >> 5;
  const int lane = tid & 31;
  const int wm   = wave >> 2;   // 0..1 : 64-row slab
  const int wn   = wave & 3;    // 0..3 : 32-col slab
  const int l16  = lane & 15;
  const int lh   = lane >> 4;   // lane-half (ISA A/C layout split)

  v8f acc[4][2];
#pragma unroll
  for (int t = 0; t < 4; ++t)
#pragma unroll
    for (int u = 0; u < 2; ++u)
      acc[t][u] = (v8f){0.f, 0.f, 0.f, 0.f, 0.f, 0.f, 0.f, 0.f};

  // Stage one BK=64 tile pair: A 128x64, B 64x128 -> 1024+1024 16B chunks,
  // 4+4 chunks (async b128 each) per thread.
  auto stage = [&](int buf, int kk) {
    bf16* la = &ldsA[buf * (128 * LDA_T)];
    bf16* lb = &ldsB[buf * (64 * LDB_T)];
#pragma unroll
    for (int i = 0; i < 4; ++i) {
      const int c  = tid + i * 256;
      const int ra = c >> 3, oa = (c & 7) * 8;
      cp16_g2l(Ab + (long)ra * lda + kk + oa, &la[ra * LDA_T + oa]);
      const int rb = c >> 4, ob = (c & 15) * 8;
      cp16_g2l(Bb + (long)(kk + rb) * ldb + ob, &lb[rb * LDB_T + ob]);
    }
  };

  stage(0, 0);
  int buf = 0;
  for (int kk = 0; kk < K; kk += 64) {
    if (kk + 64 < K) {
      stage(buf ^ 1, kk + 64);   // prefetch next tile into other buffer
      wait_async_le<8>();        // current tile's 8 async ops complete
    } else {
      wait_async_le<0>();
    }
    __syncthreads();             // all waves' staging visible

    const bf16* la = &ldsA[buf * (128 * LDA_T)];
    const bf16* lb = &ldsB[buf * (64 * LDB_T)];
#pragma unroll
    for (int s = 0; s < 2; ++s) {  // two K=32 sub-steps
      FragAB af[4];
#pragma unroll
      for (int t = 0; t < 4; ++t) {
        const bf16* p = &la[(wm * 64 + t * 16 + l16) * LDA_T + s * 32 + lh * 8];
        af[t].q[0] = *(const uint4*)(p);
        af[t].q[1] = *(const uint4*)(p + 16);
      }
      FragAB bg[2];
#pragma unroll
      for (int u = 0; u < 2; ++u) {
        const bf16* p = &lb[(s * 32 + lh * 8) * LDB_T + wn * 32 + u * 16 + l16];
#pragma unroll
        for (int e = 0; e < 8; ++e) {
          bg[u].v[e]     = p[e * LDB_T];
          bg[u].v[8 + e] = p[(16 + e) * LDB_T];
        }
      }
#pragma unroll
      for (int t = 0; t < 4; ++t)
#pragma unroll
        for (int u = 0; u < 2; ++u)
          acc[t][u] = __builtin_amdgcn_wmma_f32_16x16x32_bf16(
              false, af[t].v, false, bg[u].v, (short)0, acc[t][u], false, false);
    }
    __syncthreads();             // reads done before buffer reuse
    buf ^= 1;
  }

  // Epilogue: C 16x16 layout: VGPR r -> M = r + 8*lane_half, N = lane&15
  const long zq = z / zdiv, zr = z % zdiv;
#pragma unroll
  for (int t = 0; t < 4; ++t) {
#pragma unroll
    for (int u = 0; u < 2; ++u) {
      const long n  = n0 + wn * 32 + u * 16 + l16;
      const float bv = bias ? bias[n] : 0.f;
      const long nq = n / nbDiv, nr = n % nbDiv;
#pragma unroll
      for (int r = 0; r < 8; ++r) {
        const long  m  = m0 + wm * 64 + t * 16 + r + 8 * lh;
        const float v  = acc[t][u][r] * alpha + bv;
        const long  oi = zq * osz1 + zr * osz2 + m * osm + nq * osnb + nr * osn;
        if (Cf) Cf[oi] = v;
        if (Cb) Cb[oi] = (bf16)v;
      }
    }
  }
}

// ---------------------------------------------------------------------------
extern "C" void kernel_launch(void* const* d_in, const int* in_sizes, int n_in,
                              void* d_out, int out_size, void* d_ws, size_t ws_size,
                              hipStream_t stream) {
  (void)in_sizes; (void)n_in; (void)out_size; (void)ws_size;

  const float* q_in = (const float*)d_in[0];
  const float* k_in = (const float*)d_in[1];
  const float* v_in = (const float*)d_in[2];
  const float* i_in = (const float*)d_in[3];
  const float* W0 = (const float*)d_in[4];  const float* b0 = (const float*)d_in[5];
  const float* W1 = (const float*)d_in[6];  const float* b1 = (const float*)d_in[7];
  const float* W2 = (const float*)d_in[8];  const float* b2 = (const float*)d_in[9];
  const float* W3 = (const float*)d_in[10]; const float* b3 = (const float*)d_in[11];

  const long NQ = (long)BSZ * SQL * DMODEL;  // 33.5M elems
  const long NK = (long)BSZ * SKL * DMODEL;  // 8.39M elems

  // Workspace carve (all sizes 256B-aligned by construction):
  char* ws = (char*)d_ws;
  bf16*  bfX = (bf16*)ws;                                   // 2*NQ   (X bf16 / final Xcat)
  bf16*  bfW = (bf16*)(ws + 2 * NQ);                        // 2*NW   (current Wt bf16)
  bf16*  Qp  = (bf16*)(ws + 2 * NQ + 33554432L);            // 2*NQ   [b][h][s][d]
  bf16*  KT  = (bf16*)(ws + 4 * NQ + 33554432L);            // 2*NK   [b][h][d][s]
  bf16*  Vp  = KT + NK;                                     // 2*NK   [b][h][s][d]
  bf16*  IMp = Vp + NK;                                     // 2*NK   [b][h][s][d]
  float* Sf  = (float*)((char*)IMp + 2 * NK);               // 4*NQ   [b*h][q][512] f32
  bf16*  P   = (bf16*)((char*)Sf + 4 * NQ);                 // 2*NQ   [b*h][q][512] bf16

  const dim3 blk(256);
  const dim3 gT(64, 64);
  const dim3 gProjQ(DMODEL / 128, SQL / 128, BSZ);
  const dim3 gProjK(DMODEL / 128, SKL / 128, BSZ);
  const dim3 gAttn(SKL / 128, SQL / 128, BSZ * HEADS);

  // ---- q projection: Qp[b][h][s][d] = query @ W0^T + b0 ----
  k_cvtT_bf16<<<gT, blk, 0, stream>>>(W0, bfW, DMODEL);
  k_cvt_bf16<<<dim3((unsigned)((NQ / 4 + 255) / 256)), blk, 0, stream>>>(q_in, bfX, NQ / 4);
  k_wmma_gemm<<<gProjQ, blk, 0, stream>>>(
      bfX, DMODEL, (long)SQL * DMODEL, bfW, DMODEL, 0, b0, 1.f,
      nullptr, Qp, DMODEL,
      1, (long)HEADS * SQL * DK, 0, DK, DK, (long)SQL * DK, 1);

  // ---- k projection, stored transposed: KT[b][h][d][s] ----
  k_cvtT_bf16<<<gT, blk, 0, stream>>>(W1, bfW, DMODEL);
  k_cvt_bf16<<<dim3((unsigned)((NK / 4 + 255) / 256)), blk, 0, stream>>>(k_in, bfX, NK / 4);
  k_wmma_gemm<<<gProjK, blk, 0, stream>>>(
      bfX, DMODEL, (long)SKL * DMODEL, bfW, DMODEL, 0, b1, 1.f,
      nullptr, KT, DMODEL,
      1, (long)HEADS * SKL * DK, 0, 1, DK, (long)SKL * DK, DK);

  // ---- v projection: Vp[b][h][s][d] ----
  k_cvtT_bf16<<<gT, blk, 0, stream>>>(W2, bfW, DMODEL);
  k_cvt_bf16<<<dim3((unsigned)((NK / 4 + 255) / 256)), blk, 0, stream>>>(v_in, bfX, NK / 4);
  k_wmma_gemm<<<gProjK, blk, 0, stream>>>(
      bfX, DMODEL, (long)SKL * DMODEL, bfW, DMODEL, 0, b2, 1.f,
      nullptr, Vp, DMODEL,
      1, (long)HEADS * SKL * DK, 0, DK, DK, (long)SKL * DK, 1);

  // ---- image_feat projection: IMp[b][h][s][d] ----
  k_cvtT_bf16<<<gT, blk, 0, stream>>>(W3, bfW, DMODEL);
  k_cvt_bf16<<<dim3((unsigned)((NK / 4 + 255) / 256)), blk, 0, stream>>>(i_in, bfX, NK / 4);
  k_wmma_gemm<<<gProjK, blk, 0, stream>>>(
      bfX, DMODEL, (long)SKL * DMODEL, bfW, DMODEL, 0, b3, 1.f,
      nullptr, IMp, DMODEL,
      1, (long)HEADS * SKL * DK, 0, DK, DK, (long)SKL * DK, 1);

  // ---- scores = (q @ k^T) / sqrt(DK), then softmax over Sk ----
  const float sc = 0.04419417382415922f;  // 1/sqrt(512)
  k_wmma_gemm<<<gAttn, blk, 0, stream>>>(
      Qp, DK, (long)SQL * DK, KT, SKL, (long)DK * SKL, nullptr, sc,
      Sf, nullptr, DK,
      1, (long)SQL * SKL, 0, SKL, SKL, 0, 1);
  k_softmax512<<<dim3(BSZ * HEADS * SQL), blk, 0, stream>>>(Sf, P);

  // ---- im_attn = softmax(p_attn @ imf) over d ----
  k_wmma_gemm<<<gAttn, blk, 0, stream>>>(
      P, SKL, (long)SQL * SKL, IMp, DK, (long)SKL * DK, nullptr, 1.f,
      Sf, nullptr, SKL,
      1, (long)SQL * DK, 0, DK, DK, 0, 1);
  k_softmax512<<<dim3(BSZ * HEADS * SQL), blk, 0, stream>>>(Sf, P);

  // ---- x = im_attn @ v, written concatenated: bfX[b][s][h*512+d] ----
  k_wmma_gemm<<<gAttn, blk, 0, stream>>>(
      P, DK, (long)SQL * DK, Vp, DK, (long)SKL * DK, nullptr, 1.f,
      nullptr, bfX, DK,
      HEADS, (long)SQL * DMODEL, DK, DMODEL, DK, 0, 1);

  // ---- output projection: out = x @ W3^T + b3 (f32) ----
  k_cvtT_bf16<<<gT, blk, 0, stream>>>(W3, bfW, DMODEL);
  k_wmma_gemm<<<dim3(DMODEL / 128, (BSZ * SQL) / 128, 1), blk, 0, stream>>>(
      bfX, DMODEL, 0, bfW, DMODEL, 0, b3, 1.f,
      (float*)d_out, nullptr, DMODEL,
      1, 0, 0, DMODEL, DMODEL, 0, 1);
}
